// MultiHeadAttention_33028298506461
// MI455X (gfx1250) — matrix operand order repaired and checked by reading, not compiled, not run on hardware
//
#include <hip/hip_runtime.h>
#include <hip/hip_bf16.h>
#include <cstdint>

typedef __bf16 bhalf;
typedef __bf16 v16bf __attribute__((ext_vector_type(16)));
typedef float  v8f   __attribute__((ext_vector_type(8)));

#define BB 8
#define TT 1024
#define DD 1024
#define HH 16
#define HS 64
#define KPAD 72   // LDS row stride (elements) to avoid bank conflicts

union FragBF {
  v16bf v;
  unsigned int u[8];
};

__device__ __forceinline__ v8f wmma_bf16(const FragBF& a, const FragBF& b, v8f c) {
  // D = A(16x32 bf16) x B(32x16 bf16) + C(16x16 f32)
  return __builtin_amdgcn_wmma_f32_16x16x32_bf16(false, a.v, false, b.v, (short)0, c,
                                                 false, false);
}

// Scheduling fence: keep batched loads as one clause ahead of the WMMA burst.
__device__ __forceinline__ void sched_fence() { __builtin_amdgcn_sched_barrier(0); }

// CDNA5 async DMA: copy 16B/lane from global memory into LDS (ASYNCcnt-tracked).
__device__ __forceinline__ void async_copy_b128(void* lds_dst, const void* gsrc) {
  unsigned lofs = (unsigned)(uintptr_t)lds_dst;  // LDS aperture: low 32 bits = LDS offset
  unsigned long long ga = (unsigned long long)(uintptr_t)gsrc;
  asm volatile("global_load_async_to_lds_b128 %0, %1, off"
               :: "v"(lofs), "v"(ga)
               : "memory");
}

__device__ __forceinline__ void wait_async0() {
  asm volatile("s_wait_asynccnt 0x0" ::: "memory");
}

// ---------------- conversion kernels ----------------

__global__ void k_cvt_f32_bf16(const float* __restrict__ src, bhalf* __restrict__ dst, int n) {
  int i = blockIdx.x * blockDim.x + threadIdx.x;
  if (i < n) dst[i] = (bhalf)src[i];
}

// Wq/Wk/Wv [H, D, HS] fp32  ->  Bt layout [N=H*HS][K=D] bf16
__global__ void k_cvt_wqkv(const float* __restrict__ src, bhalf* __restrict__ dst) {
  int i = blockIdx.x * blockDim.x + threadIdx.x;
  if (i >= DD * HH * HS) return;
  int c = i >> 10;        // output column = h*HS + s
  int d = i & (DD - 1);   // K index
  int h = c >> 6;
  int s = c & (HS - 1);
  dst[i] = (bhalf)src[(h * DD + d) * HS + s];
}

// ---------------- generic WMMA GEMM (register double-buffered) ----------------
// C[MxN] = A[MxK] (bf16 row-major) x Bt[NxK]^T (bf16, B stored transposed)
// OUTMODE 0: fp32 out + bias; 1: bf16 row-major; 2: bf16 [B,H,HS,T] (for V)
template <int OUTMODE>
__global__ __launch_bounds__(128) void k_gemm_bf16(
    const bhalf* __restrict__ A, const bhalf* __restrict__ Bt,
    float* __restrict__ Cf, bhalf* __restrict__ Cb, const float* __restrict__ bias,
    int M, int N, int K) {
  const int lane = threadIdx.x & 31;
  const int w    = threadIdx.x >> 5;
  const int mn   = lane & 15;   // M (for A) / N (for B & C) lane index
  const int kh   = lane >> 4;   // half-wave selector
  const int r0   = blockIdx.y * 128 + w * 32;
  const int c0   = blockIdx.x * 64;

  const v8f zero8 = {0.f, 0.f, 0.f, 0.f, 0.f, 0.f, 0.f, 0.f};
  v8f acc[2][4];
#pragma unroll
  for (int i = 0; i < 2; ++i)
#pragma unroll
    for (int j = 0; j < 4; ++j) acc[i][j] = zero8;

  auto loadA = [&](FragBF a[2], int k0) {
#pragma unroll
    for (int mi = 0; mi < 2; ++mi)
#pragma unroll
      for (int v = 0; v < 8; ++v) {
        // A 16x32 bf16 layout: VGPR quad selects K 16-chunk, lane-half selects 8-chunk
        int koff = ((v & 4) << 2) + kh * 8 + ((v & 3) << 1);
        a[mi].u[v] = *(const unsigned int*)(A + (size_t)(r0 + mi * 16 + mn) * K + k0 + koff);
      }
  };
  auto loadB = [&](FragBF bfr[4], int k0) {
#pragma unroll
    for (int ct = 0; ct < 4; ++ct)
#pragma unroll
      for (int v = 0; v < 8; ++v)
        bfr[ct].u[v] = *(const unsigned int*)(Bt + (size_t)(c0 + ct * 16 + mn) * K +
                                              k0 + ((v & 4) << 2) + kh * 8 + ((v & 3) << 1));
  };

  FragBF a0[2], b0[4], a1[2], b1[4];
  loadA(a0, 0);
  loadB(b0, 0);

  for (int k0 = 0; k0 < K; k0 += 64) {
    // prefetch second half of this 64-chunk while multiplying the first
    loadA(a1, k0 + 32);
    loadB(b1, k0 + 32);
    sched_fence();
#pragma unroll
    for (int ct = 0; ct < 4; ++ct)
#pragma unroll
      for (int mi = 0; mi < 2; ++mi) acc[mi][ct] = wmma_bf16(a0[mi], b0[ct], acc[mi][ct]);
    sched_fence();

    // prefetch next 64-chunk (clamped in-bounds; values unused on last iter)
    const int kn = (k0 + 64 < K) ? (k0 + 64) : 0;
    loadA(a0, kn);
    loadB(b0, kn);
    sched_fence();
#pragma unroll
    for (int ct = 0; ct < 4; ++ct)
#pragma unroll
      for (int mi = 0; mi < 2; ++mi) acc[mi][ct] = wmma_bf16(a1[mi], b1[ct], acc[mi][ct]);
    sched_fence();
  }

#pragma unroll
  for (int mi = 0; mi < 2; ++mi)
#pragma unroll
    for (int ct = 0; ct < 4; ++ct)
#pragma unroll
      for (int v = 0; v < 8; ++v) {
        int row = r0 + mi * 16 + kh * 8 + v;  // C layout: M = v + 8*lane_half
        int col = c0 + ct * 16 + mn;
        float val = acc[mi][ct][v];
        if (OUTMODE == 0) {
          Cf[(size_t)row * N + col] = val + bias[col];
        } else if (OUTMODE == 1) {
          Cb[(size_t)row * N + col] = (bhalf)val;
        } else {
          // [B,H,HS,T]: row = b*T + t, col = h*HS + s
          Cb[((size_t)((row >> 10) * HH + (col >> 6)) * HS + (col & (HS - 1))) * TT +
             (row & (TT - 1))] = (bhalf)val;
        }
      }
}

// ---------------- flash attention (causal) ----------------
// Q,K: bf16 [B*T, D] (col = h*HS + hs); Vt: bf16 [B,H,HS,T]; O: bf16 [B*T, D]
// K/V tiles are staged into double-buffered LDS with async-DMA, prefetching
// tile jb+1 while computing tile jb.
__global__ __launch_bounds__(128) void k_attn(
    const bhalf* __restrict__ Q, const bhalf* __restrict__ Kb,
    const bhalf* __restrict__ Vt, bhalf* __restrict__ O) {
  __shared__ bhalf kbuf[2][64 * KPAD];  // [key][hs]
  __shared__ bhalf vbuf[2][64 * KPAD];  // [hs][key]
  __shared__ bhalf pl[4][16 * KPAD];    // per-wave P staging (C->A layout change)

  const int tid  = threadIdx.x;
  const int lane = tid & 31;
  const int w    = tid >> 5;
  const int mn   = lane & 15;
  const int kh   = lane >> 4;
  const int b  = blockIdx.z;
  const int h  = blockIdx.y;
  const int tb = blockIdx.x;
  const int tq0 = tb * 64 + w * 16;     // this wave's 16 query rows
  const size_t rb = (size_t)b * TT;

  // Q A-fragments (16 rows x 64 HS = 2 K-chunks), loaded once
  FragBF qf[2];
#pragma unroll
  for (int kc = 0; kc < 2; ++kc)
#pragma unroll
    for (int v = 0; v < 8; ++v) {
      int koff = ((v & 4) << 2) + kh * 8 + ((v & 3) << 1);
      qf[kc].u[v] = *(const unsigned int*)(Q + (rb + tq0 + mn) * DD + h * HS + kc * 32 + koff);
    }

  // stage K/V tile jb into LDS buffer bf: 128 threads x 4 x 16B = 8KB per tile
  auto stage = [&](int jb, int bf) {
    const int j0 = jb * 64;
#pragma unroll
    for (int i = 0; i < 4; ++i) {
      const int idx = tid + i * 128;    // 0..511
      const int row = idx >> 3;         // 0..63
      const int seg = idx & 7;          // 16B segment within 128B row
      async_copy_b128(&kbuf[bf][row * KPAD + seg * 8],
                      Kb + (rb + j0 + row) * DD + h * HS + seg * 8);
      async_copy_b128(&vbuf[bf][row * KPAD + seg * 8],
                      Vt + ((size_t)((b * HH + h) * HS + row)) * TT + j0 + seg * 8);
    }
  };

  stage(0, 0);
  wait_async0();
  __syncthreads();

  const v8f zero8 = {0.f, 0.f, 0.f, 0.f, 0.f, 0.f, 0.f, 0.f};
  v8f o[4];
#pragma unroll
  for (int ct = 0; ct < 4; ++ct) o[ct] = zero8;
  float m[8], l[8];
#pragma unroll
  for (int v = 0; v < 8; ++v) { m[v] = -1e30f; l[v] = 0.f; }

  for (int jb = 0; jb <= tb; ++jb) {
    const int j0  = jb * 64;
    const int cur = jb & 1;
    if (jb < tb) stage(jb + 1, cur ^ 1);  // async prefetch next tile

    // S(16x64) = Q(16x64) @ K_blk^T(64x64): batch all 8 B-frags, then 8 WMMAs
    FragBF bk[8];
#pragma unroll
    for (int ct = 0; ct < 4; ++ct)
#pragma unroll
      for (int kc = 0; kc < 2; ++kc)
#pragma unroll
        for (int v = 0; v < 8; ++v)
          bk[ct * 2 + kc].u[v] = *(const unsigned int*)
              &kbuf[cur][(ct * 16 + mn) * KPAD + kc * 32 + ((v & 4) << 2) + kh * 8 + ((v & 3) << 1)];
    sched_fence();  // all 16 ds_load_b128 issue as a clause ahead of the WMMAs
    v8f s[4];
#pragma unroll
    for (int ct = 0; ct < 4; ++ct) s[ct] = zero8;
#pragma unroll
    for (int ct = 0; ct < 4; ++ct)
#pragma unroll
      for (int kc = 0; kc < 2; ++kc) s[ct] = wmma_bf16(qf[kc], bk[ct * 2 + kc], s[ct]);

    // scale + causal mask + online softmax (row = kh*8 + v in C layout)
#pragma unroll
    for (int v = 0; v < 8; ++v) {
      const int trow = tq0 + kh * 8 + v;
      float sv[4];
      float rm = -1e30f;
#pragma unroll
      for (int ct = 0; ct < 4; ++ct) {
        int key = j0 + ct * 16 + mn;
        float val = s[ct][v] * 0.125f;  // 1/sqrt(HS)
        if (key > trow) val = -1e30f;
        sv[ct] = val;
        rm = fmaxf(rm, val);
      }
#pragma unroll
      for (int d = 1; d < 16; d <<= 1) rm = fmaxf(rm, __shfl_xor(rm, d, 32));
      float mnew  = fmaxf(m[v], rm);
      float alpha = __expf(m[v] - mnew);
      float rs = 0.f;
#pragma unroll
      for (int ct = 0; ct < 4; ++ct) {
        float pv = __expf(sv[ct] - mnew);
        pl[w][(kh * 8 + v) * KPAD + ct * 16 + mn] = (bhalf)pv;
        rs += pv;
        o[ct][v] *= alpha;
      }
#pragma unroll
      for (int d = 1; d < 16; d <<= 1) rs += __shfl_xor(rs, d, 32);
      l[v] = l[v] * alpha + rs;
      m[v] = mnew;
    }

    // O += P(16x64) @ V_blk(64x64); P read back in A-fragment layout (same wave)
    FragBF pf[2];
#pragma unroll
    for (int kc = 0; kc < 2; ++kc)
#pragma unroll
      for (int v = 0; v < 8; ++v) {
        int koff = ((v & 4) << 2) + kh * 8 + ((v & 3) << 1);
        pf[kc].u[v] = *(const unsigned int*)&pl[w][mn * KPAD + kc * 32 + koff];
      }
    FragBF bv[8];
#pragma unroll
    for (int ct = 0; ct < 4; ++ct)
#pragma unroll
      for (int kc = 0; kc < 2; ++kc)
#pragma unroll
        for (int v = 0; v < 8; ++v)
          bv[ct * 2 + kc].u[v] = *(const unsigned int*)
              &vbuf[cur][(ct * 16 + mn) * KPAD + kc * 32 + ((v & 4) << 2) + kh * 8 + ((v & 3) << 1)];
    sched_fence();  // P + V loads clause together ahead of the WMMA burst
#pragma unroll
    for (int ct = 0; ct < 4; ++ct)
#pragma unroll
      for (int kc = 0; kc < 2; ++kc) o[ct] = wmma_bf16(pf[kc], bv[ct * 2 + kc], o[ct]);

    // prefetched tile must be complete & visible before next iteration reads it
    wait_async0();
    __syncthreads();
  }

  // finalize: divide by row sums, write bf16 [B*T, D]
#pragma unroll
  for (int v = 0; v < 8; ++v) {
    float inv = 1.0f / l[v];
#pragma unroll
    for (int ct = 0; ct < 4; ++ct)
      O[(rb + tq0 + kh * 8 + v) * DD + h * HS + ct * 16 + mn] = (bhalf)(o[ct][v] * inv);
  }
}

// ---------------- launcher ----------------

extern "C" void kernel_launch(void* const* d_in, const int* in_sizes, int n_in,
                              void* d_out, int out_size, void* d_ws, size_t ws_size,
                              hipStream_t stream) {
  (void)in_sizes; (void)n_in; (void)out_size; (void)ws_size;
  const float* x  = (const float*)d_in[0];
  const float* Wq = (const float*)d_in[1];
  const float* Wk = (const float*)d_in[2];
  const float* Wv = (const float*)d_in[3];
  const float* Wo = (const float*)d_in[4];
  const float* bo = (const float*)d_in[5];
  float* out = (float*)d_out;

  char* ws = (char*)d_ws;
  const size_t MB = 1u << 20;
  bhalf* xb  = (bhalf*)(ws + 0 * MB);    // 16 MB: x bf16 [8192,1024]
  bhalf* wqt = (bhalf*)(ws + 16 * MB);   //  2 MB: Wq Bt [1024,1024]
  bhalf* wkt = (bhalf*)(ws + 18 * MB);
  bhalf* wvt = (bhalf*)(ws + 20 * MB);
  bhalf* wot = (bhalf*)(ws + 22 * MB);   //  2 MB: Wo (already Bt for x@Wo^T)
  bhalf* qb  = (bhalf*)(ws + 24 * MB);   // 16 MB: Q [8192,1024]
  bhalf* kb  = (bhalf*)(ws + 40 * MB);   // 16 MB: K [8192,1024]
  bhalf* vbt = (bhalf*)(ws + 56 * MB);   // 16 MB: V [B,H,HS,T]
  bhalf* ab  = (bhalf*)(ws + 72 * MB);   // 16 MB: attention out [8192,1024]

  const int NX = BB * TT * DD;  // 8M
  const int NW = DD * HH * HS;  // 1M
  k_cvt_f32_bf16<<<(NX + 255) / 256, 256, 0, stream>>>(x, xb, NX);
  k_cvt_wqkv<<<(NW + 255) / 256, 256, 0, stream>>>(Wq, wqt);
  k_cvt_wqkv<<<(NW + 255) / 256, 256, 0, stream>>>(Wk, wkt);
  k_cvt_wqkv<<<(NW + 255) / 256, 256, 0, stream>>>(Wv, wvt);
  k_cvt_f32_bf16<<<(NW + 255) / 256, 256, 0, stream>>>(Wo, wot, NW);

  dim3 gg(DD / 64, (BB * TT) / 128);
  k_gemm_bf16<1><<<gg, 128, 0, stream>>>(xb, wqt, nullptr, qb, nullptr, BB * TT, DD, DD);
  k_gemm_bf16<1><<<gg, 128, 0, stream>>>(xb, wkt, nullptr, kb, nullptr, BB * TT, DD, DD);
  k_gemm_bf16<2><<<gg, 128, 0, stream>>>(xb, wvt, nullptr, vbt, nullptr, BB * TT, DD, DD);

  dim3 ga(TT / 64, HH, BB);
  k_attn<<<ga, 128, 0, stream>>>(qb, kb, vbt, ab);

  k_gemm_bf16<0><<<gg, 128, 0, stream>>>(ab, wot, out, nullptr, bo, BB * TT, DD, DD);
}
